// MoEBlock_61400852464237
// MI455X (gfx1250) — matrix operand order repaired and checked by reading, not compiled, not run on hardware
//
#include <hip/hip_runtime.h>
#include <hip/hip_bf16.h>

typedef __attribute__((ext_vector_type(16))) __bf16 v16bf;
typedef __attribute__((ext_vector_type(8)))  float  v8f;

#define T_TOK 32768
#define E_EXP 64
#define K_DIM 1024
#define I_DIM 512
#define CAP   1024
#define MT    32           // token rows per block
#define BT_STRIDE 40       // bf16 elems per Bt row: 32 K + 8 pad (16B-aligned frags, no bank conflicts)
#define H_STRIDE  520      // bf16 elems per h row: 512 + 8 pad

static __device__ __forceinline__ unsigned short bf16bits(float f) {
    return __builtin_bit_cast(unsigned short, (__bf16)f);
}

// silu(g)*u with hardware rcp (v_rcp_f32) instead of IEEE divide
static __device__ __forceinline__ float silu_mul(float g, float u) {
    float sig = __builtin_amdgcn_rcpf(1.0f + __expf(-g));
    return g * sig * u;
}

static __device__ __forceinline__ v16bf load_bfrag(const unsigned char* sbt, int n, int hsel) {
    // Bt[n][kk], lane reads 8 bf16 at kk = hsel*8 and 8 bf16 at kk = 16 + hsel*8
    v16bf b;
    __builtin_memcpy(&b, sbt + ((size_t)n * BT_STRIDE + hsel * 8) * 2, 16);
    __builtin_memcpy(((char*)&b) + 16, sbt + ((size_t)n * BT_STRIDE + 16 + hsel * 8) * 2, 16);
    return b;
}

__global__ __launch_bounds__(256, 1)
void moe_fused_kernel(const float* __restrict__ x,
                      const int*   __restrict__ tpe,
                      const float* __restrict__ w13,
                      const float* __restrict__ w2,
                      float* __restrict__ out)
{
    __shared__ __align__(16) unsigned char s_h [MT * H_STRIDE * 2];   // 33280 B  bf16 h[32][512+pad]
    __shared__ __align__(16) unsigned char s_bt[256 * BT_STRIDE * 2]; // 20480 B  bf16 Bt[256][32+pad]

    const int e     = blockIdx.y;
    const int mtile = blockIdx.x;

    int start = 0;
    for (int i = 0; i < E_EXP; ++i) { int c = tpe[i]; start += (i < e) ? c : 0; }
    const int cnt  = tpe[e];
    const int row0 = mtile * MT;
    if (row0 >= cnt) return;                       // uniform across block
    const int rows_valid = min(MT, cnt - row0);

    const int tid  = threadIdx.x;
    const int wave = tid >> 5;
    const int lane = tid & 31;
    const int hsel = lane >> 4;                    // lane group 0/1
    const int lcol = lane & 15;

    const int mg = wave & 1;                       // 16-row group
    const int cg = wave >> 1;                      // 0..3 : 32-col group

    // ---- Phase 1 A source: x row (clamped to stay in-bounds for padded rows) ----
    int grow = start + row0 + mg * 16 + lcol;
    if (grow >= T_TOK) grow = T_TOK - 1;
    const float* xrow = x + (size_t)grow * K_DIM;
    const float* w13e = w13 + (size_t)e * K_DIM * (2 * I_DIM);

    const int btn = tid;                           // this thread owns Bt row `btn`

    // =================== Phase 1: gate_up = x @ w13, SwiGLU -> h (LDS) ===================
    for (int j = 0; j < 4; ++j) {                  // 128 gate cols + 128 up cols per chunk
        v8f a0 = {}, a1 = {}, a2 = {}, a3 = {};
        const int colg = (btn < 128) ? (j * 128 + btn)
                                     : (I_DIM + j * 128 + (btn - 128));
        const float* wp = w13e + colg;

        for (int kb = 0; kb < K_DIM; kb += 32) {
            __syncthreads();
            // stage transposed bf16 B tile: Bt[btn][kk] = w13[kb+kk][colg]
            #pragma unroll
            for (int kk = 0; kk < 32; kk += 2) {
                float g0 = wp[(size_t)(kb + kk)     * (2 * I_DIM)];
                float g1 = wp[(size_t)(kb + kk + 1) * (2 * I_DIM)];
                unsigned int p = (unsigned int)bf16bits(g0) | ((unsigned int)bf16bits(g1) << 16);
                *(unsigned int*)(s_bt + ((size_t)btn * BT_STRIDE + kk) * 2) = p;
            }
            __syncthreads();

            // A fragment: per-lane K runs {hsel*8..+7} and {16+hsel*8..+7}
            const int klo = kb + hsel * 8;
            const int khi = kb + 16 + hsel * 8;
            float4 f0 = *(const float4*)(xrow + klo);
            float4 f1 = *(const float4*)(xrow + klo + 4);
            float4 f2 = *(const float4*)(xrow + khi);
            float4 f3 = *(const float4*)(xrow + khi + 4);
            v16bf a;
            a[0]=(__bf16)f0.x; a[1]=(__bf16)f0.y; a[2]=(__bf16)f0.z; a[3]=(__bf16)f0.w;
            a[4]=(__bf16)f1.x; a[5]=(__bf16)f1.y; a[6]=(__bf16)f1.z; a[7]=(__bf16)f1.w;
            a[8]=(__bf16)f2.x; a[9]=(__bf16)f2.y; a[10]=(__bf16)f2.z; a[11]=(__bf16)f2.w;
            a[12]=(__bf16)f3.x; a[13]=(__bf16)f3.y; a[14]=(__bf16)f3.z; a[15]=(__bf16)f3.w;

            const int nb = cg * 32 + lcol;
            v16bf b;
            b = load_bfrag(s_bt, nb,            hsel);
            a0 = __builtin_amdgcn_wmma_f32_16x16x32_bf16(false, a, false, b, (short)0, a0, false, false);
            b = load_bfrag(s_bt, nb + 16,       hsel);
            a1 = __builtin_amdgcn_wmma_f32_16x16x32_bf16(false, a, false, b, (short)0, a1, false, false);
            b = load_bfrag(s_bt, 128 + nb,      hsel);
            a2 = __builtin_amdgcn_wmma_f32_16x16x32_bf16(false, a, false, b, (short)0, a2, false, false);
            b = load_bfrag(s_bt, 128 + nb + 16, hsel);
            a3 = __builtin_amdgcn_wmma_f32_16x16x32_bf16(false, a, false, b, (short)0, a3, false, false);
        }

        // SwiGLU in registers, store bf16 h to LDS (gate = a0/a1, up = a2/a3, same cols)
        const int hc = j * 128 + cg * 32 + lcol;
        #pragma unroll
        for (int v = 0; v < 8; ++v) {
            const int r = mg * 16 + v + 8 * hsel;
            *(unsigned short*)(s_h + ((size_t)r * H_STRIDE + hc) * 2)      = bf16bits(silu_mul(a0[v], a2[v]));
            *(unsigned short*)(s_h + ((size_t)r * H_STRIDE + hc + 16) * 2) = bf16bits(silu_mul(a1[v], a3[v]));
        }
    }

    // =================== Phase 2: out = h @ w2 ===================
    const int mg2 = wave & 1;
    const int cg2 = wave >> 1;                     // 0..3 : 64-col group
    const float* w2e = w2 + (size_t)e * I_DIM * K_DIM;

    for (int j2 = 0; j2 < 4; ++j2) {               // 256 out cols per chunk
        v8f a0 = {}, a1 = {}, a2 = {}, a3 = {};
        const float* wp2 = w2e + (j2 * 256 + btn);

        for (int ib = 0; ib < I_DIM; ib += 32) {
            __syncthreads();
            #pragma unroll
            for (int ii = 0; ii < 32; ii += 2) {
                float g0 = wp2[(size_t)(ib + ii)     * K_DIM];
                float g1 = wp2[(size_t)(ib + ii + 1) * K_DIM];
                unsigned int p = (unsigned int)bf16bits(g0) | ((unsigned int)bf16bits(g1) << 16);
                *(unsigned int*)(s_bt + ((size_t)btn * BT_STRIDE + ii) * 2) = p;
            }
            __syncthreads();

            // A fragment from h in LDS
            const int m   = mg2 * 16 + lcol;
            const int ilo = ib + hsel * 8;
            const int ihi = ib + 16 + hsel * 8;
            v16bf a;
            __builtin_memcpy(&a,                 s_h + ((size_t)m * H_STRIDE + ilo) * 2, 16);
            __builtin_memcpy(((char*)&a) + 16,   s_h + ((size_t)m * H_STRIDE + ihi) * 2, 16);

            const int nb = cg2 * 64 + lcol;
            v16bf b;
            b = load_bfrag(s_bt, nb,      hsel);
            a0 = __builtin_amdgcn_wmma_f32_16x16x32_bf16(false, a, false, b, (short)0, a0, false, false);
            b = load_bfrag(s_bt, nb + 16, hsel);
            a1 = __builtin_amdgcn_wmma_f32_16x16x32_bf16(false, a, false, b, (short)0, a1, false, false);
            b = load_bfrag(s_bt, nb + 32, hsel);
            a2 = __builtin_amdgcn_wmma_f32_16x16x32_bf16(false, a, false, b, (short)0, a2, false, false);
            b = load_bfrag(s_bt, nb + 48, hsel);
            a3 = __builtin_amdgcn_wmma_f32_16x16x32_bf16(false, a, false, b, (short)0, a3, false, false);
        }

        // store f32 output rows (only valid token rows)
        #pragma unroll
        for (int v = 0; v < 8; ++v) {
            const int r = mg2 * 16 + v + 8 * hsel;
            if (r < rows_valid) {
                float* orow = out + (size_t)(start + row0 + r) * K_DIM;
                const int c0 = j2 * 256 + cg2 * 64 + lcol;
                orow[c0]      = a0[v];
                orow[c0 + 16] = a1[v];
                orow[c0 + 32] = a2[v];
                orow[c0 + 48] = a3[v];
            }
        }
    }
}

extern "C" void kernel_launch(void* const* d_in, const int* in_sizes, int n_in,
                              void* d_out, int out_size, void* d_ws, size_t ws_size,
                              hipStream_t stream) {
    const float* x   = (const float*)d_in[0];
    const int*   tpe = (const int*)  d_in[1];
    const float* w13 = (const float*)d_in[2];
    const float* w2  = (const float*)d_in[3];
    float* out = (float*)d_out;

    dim3 grid(CAP / MT, E_EXP);   // 32 row-tiles x 64 experts; over-capacity tiles self-exit
    moe_fused_kernel<<<grid, 256, 0, stream>>>(x, tpe, w13, w2, out);
}
